// VTA_24885040513563
// MI455X (gfx1250) — matrix operand order repaired and checked by reading, not compiled, not run on hardware
//
#include <hip/hip_runtime.h>
#include <hip/hip_bf16.h>

typedef __attribute__((ext_vector_type(16))) __bf16           v16bf;
typedef __attribute__((ext_vector_type(8)))  float            v8f;
typedef __attribute__((ext_vector_type(8)))  unsigned short   us8;
typedef __attribute__((ext_vector_type(16))) unsigned short   us16;

static __device__ __forceinline__ unsigned short f2bf(float f) {
  unsigned u = __float_as_uint(f);
  unsigned r = (u + 0x7FFFu + ((u >> 16) & 1u)) >> 16;
  return (unsigned short)r;
}
static __device__ __forceinline__ float sigmoidf_(float x) { return 1.f / (1.f + __expf(-x)); }
static __device__ __forceinline__ float siluf_(float x)    { return x * sigmoidf_(x); }

// ---------------------------------------------------------------------------
// bf16 WMMA GEMM: Y[M,N] = X[M,K] @ W^T  (W is [N,K] row-major bf16).
// Requirements (guaranteed by call sites): K % 64 == 0, N % 64 == 0, M % 16 == 0.
// ntaps==3 => zero-padded centred conv3 over time (rows are (b,t), shift = tap-1),
// weight blocks of [N,K] per tap contiguous.
// Wave owns 4 N-tiles (64 cols); A-tile double-buffered in LDS (1 barrier/chunk).
// All 8 B fragments of a chunk are loaded into distinct registers BEFORE any
// WMMA so the scheduler can clause the loads and issue the 8 WMMAs back-to-back.
// ---------------------------------------------------------------------------
__global__ __launch_bounds__(256) void gemm_bf16_k(
    const float* __restrict__ X, const unsigned short* __restrict__ W,
    float* __restrict__ Y, int M, int N, int K, int ldx, int ntaps, int Tdim)
{
  __shared__ __align__(16) unsigned short As[2][16 * 64];
  const int tid  = threadIdx.x;
  const int wave = tid >> 5;
  const int lane = tid & 31;
  const int lm   = lane & 15;
  const int lh   = lane >> 4;
  const int mtile = blockIdx.x;
  const int n0    = (blockIdx.y * 8 + wave) * 64;  // 4 tiles: n0 .. n0+48
  const int sr = tid >> 4;          // staging row 0..15
  const int sc = (tid & 15) << 2;   // staging col base 0..60
  const bool active = (n0 < N);     // uniform per wave; N % 64 == 0

  v8f acc[4];
#pragma unroll
  for (int j = 0; j < 4; ++j)
#pragma unroll
    for (int i = 0; i < 8; ++i) acc[j][i] = 0.f;

  size_t rowoff[4];
#pragma unroll
  for (int j = 0; j < 4; ++j)
    rowoff[j] = (size_t)(n0 + j * 16 + lm) * (size_t)K + (size_t)(lh * 8);

  auto stage_load = [&](int tap, int k0) -> float4 {
    float4 v = make_float4(0.f, 0.f, 0.f, 0.f);
    int m = mtile * 16 + sr;
    int shift = 0;
    bool valid = true;
    if (ntaps == 3) { shift = tap - 1; int tt = (m % Tdim) + shift; valid = (tt >= 0) && (tt < Tdim); }
    if (valid) v = *(const float4*)(X + (size_t)(m + shift) * ldx + k0 + sc);
    return v;
  };
  auto stage_store = [&](unsigned short* buf, float4 v) {
    buf[sr * 64 + sc + 0] = f2bf(v.x);
    buf[sr * 64 + sc + 1] = f2bf(v.y);
    buf[sr * 64 + sc + 2] = f2bf(v.z);
    buf[sr * 64 + sc + 3] = f2bf(v.w);
  };
  auto loadB16 = [&](const unsigned short* p) -> us16 {
    us8 x0 = *(const us8*)p;
    us8 x1 = *(const us8*)(p + 16);
    us16 r;
#pragma unroll
    for (int i = 0; i < 8; ++i) { r[i] = x0[i]; r[8 + i] = x1[i]; }
    return r;
  };

  const int nch   = K >> 6;          // K/64 chunks per tap
  const int total = ntaps * nch;

  // prologue: stage chunk 0 into buffer 0
  stage_store(As[0], stage_load(0, 0));
  __syncthreads();

  for (int c = 0; c < total; ++c) {
    const int tap = (ntaps == 3) ? (c / nch) : 0;
    const int k0  = (c - tap * nch) << 6;
    const bool has = (c + 1 < total);
    float4 vn;
    if (has) {
      const int tn = (ntaps == 3) ? ((c + 1) / nch) : 0;
      const int kn = ((c + 1) - tn * nch) << 6;
      vn = stage_load(tn, kn);  // issue early; latency hidden by WMMAs below
      if (active)
        __builtin_prefetch(W + (size_t)tn * N * K + rowoff[0] + kn, 0, 1);
    }

    if (active) {
      const unsigned short* ab = As[c & 1];
      const unsigned short* wk = W + (size_t)tap * N * K + k0;

      // Both A fragments (kk=0,1): 4x ds_load_b128.
      // Lane(m=lm, half=lh) holds K in [8h,8h+8) and [16+8h,16+8h+8) per kk.
      v16bf avv[2];
#pragma unroll
      for (int kk = 0; kk < 2; ++kk) {
        us8 g0 = *(const us8*)&ab[lm * 64 + kk * 32 + lh * 8];
        us8 g1 = *(const us8*)&ab[lm * 64 + kk * 32 + 16 + lh * 8];
        us16 au;
#pragma unroll
        for (int i = 0; i < 8; ++i) { au[i] = g0[i]; au[8 + i] = g1[i]; }
        avv[kk] = __builtin_bit_cast(v16bf, au);
      }

      // All 8 B fragments into distinct registers first (16x global_load_b128,
      // clause-able), then 8 WMMAs back-to-back on independent accumulators.
      us16 bf[8];
#pragma unroll
      for (int s = 0; s < 8; ++s)
        bf[s] = loadB16(wk + (s >> 2) * 32 + rowoff[s & 3]);
#pragma unroll
      for (int s = 0; s < 8; ++s) {
        v16bf bv = __builtin_bit_cast(v16bf, bf[s]);
        acc[s & 3] = __builtin_amdgcn_wmma_f32_16x16x32_bf16(false, avv[s >> 2], false, bv,
                                                             (short)0, acc[s & 3], false, false);
      }
    }

    if (has) stage_store(As[(c + 1) & 1], vn);
    __syncthreads();
  }

  if (active) {
    // D layout: element v of lane L -> row mtile*16 + 8*(L/16)+v, col n0+(L%16)
    float* yr = Y + (size_t)(mtile * 16 + lh * 8) * N + n0 + lm;
#pragma unroll
    for (int j = 0; j < 4; ++j)
#pragma unroll
      for (int i = 0; i < 8; ++i) yr[(size_t)i * N + j * 16] = acc[j][i];
  }
}

// ---------------- LayerNorm (+optional SiLU), one block per row -------------
__global__ __launch_bounds__(256) void ln_act_k(const float* __restrict__ X,
    float* __restrict__ Y, const float* __restrict__ g, const float* __restrict__ b,
    int N, int act)
{
  __shared__ float s1[256], s2[256];
  const int row = blockIdx.x, tid = threadIdx.x;
  const float* xr = X + (size_t)row * N;
  float sum = 0.f, sq = 0.f;
  for (int c = tid; c < N; c += 256) { float x = xr[c]; sum += x; sq += x * x; }
  s1[tid] = sum; s2[tid] = sq; __syncthreads();
  for (int o = 128; o > 0; o >>= 1) {
    if (tid < o) { s1[tid] += s1[tid + o]; s2[tid] += s2[tid + o]; }
    __syncthreads();
  }
  float mean = s1[0] / N;
  float var  = s2[0] / N - mean * mean; var = var < 0.f ? 0.f : var;
  float r = rsqrtf(var + 0.001f);
  float* yr = Y + (size_t)row * N;
  for (int c = tid; c < N; c += 256) {
    float y = (xr[c] - mean) * r * g[c] + b[c];
    yr[c] = act ? siluf_(y) : y;
  }
}

// ---------------- BatchNorm stats (per channel over M rows) -----------------
__global__ __launch_bounds__(256) void bn_stats_k(const float* __restrict__ X,
    float* __restrict__ mean, float* __restrict__ var, int Mrows, int N)
{
  __shared__ float s1[256], s2[256];
  const int ch = blockIdx.x, tid = threadIdx.x;
  float sum = 0.f, sq = 0.f;
  for (int m = tid; m < Mrows; m += 256) {
    float x = X[(size_t)m * N + ch]; sum += x; sq += x * x;
  }
  s1[tid] = sum; s2[tid] = sq; __syncthreads();
  for (int o = 128; o > 0; o >>= 1) {
    if (tid < o) { s1[tid] += s1[tid + o]; s2[tid] += s2[tid + o]; }
    __syncthreads();
  }
  if (tid == 0) {
    float mu = s1[0] / Mrows;
    float v  = s2[0] / Mrows - mu * mu;
    mean[ch] = mu; var[ch] = v < 0.f ? 0.f : v;
  }
}

__global__ __launch_bounds__(256) void bn_silu_k(float* __restrict__ X,
    const float* __restrict__ mean, const float* __restrict__ var,
    const float* __restrict__ g, const float* __restrict__ b, int total, int N)
{
  int i = blockIdx.x * 256 + threadIdx.x;
  if (i >= total) return;
  int c = i % N;
  float x = (X[i] - mean[c]) * rsqrtf(var[c] + 1e-5f) * g[c] + b[c];
  X[i] = siluf_(x);
}

// ---------------- tiny conv head: N=2 logits ---------------------------------
__global__ __launch_bounds__(128) void pbout_k(const float* __restrict__ X,
    const float* __restrict__ w, const float* __restrict__ bias,
    float* __restrict__ LA, int Tdim, int K)
{
  __shared__ float sA[128], sB[128];
  const int row = blockIdx.x, tid = threadIdx.x;
  const int t = row % Tdim;
  float a0 = 0.f, a1 = 0.f;
  for (int s = 0; s < 3; ++s) {
    int ts = t + s - 1;
    if (ts < 0 || ts >= Tdim) continue;
    const float* xr = X + (size_t)(row + s - 1) * K;
    for (int i = tid; i < K; i += 128) {
      float x = xr[i];
      a0 += x * w[(size_t)i * 3 + s];
      a1 += x * w[(size_t)K * 3 + (size_t)i * 3 + s];
    }
  }
  sA[tid] = a0; sB[tid] = a1; __syncthreads();
  for (int o = 64; o > 0; o >>= 1) {
    if (tid < o) { sA[tid] += sA[tid + o]; sB[tid] += sB[tid + o]; }
    __syncthreads();
  }
  if (tid == 0) { LA[(size_t)row * 2 + 0] = sA[0] + bias[0]; LA[(size_t)row * 2 + 1] = sB[0] + bias[1]; }
}

// ---------------- Gumbel gating; read == hard argmax under ST estimator -----
__global__ void gate_k(const float* __restrict__ LA, const float* __restrict__ U,
    float* __restrict__ segL, float* __restrict__ segN, float* __restrict__ readv,
    int t, int Tdim, int B)
{
  int b = blockIdx.x * blockDim.x + threadIdx.x;
  if (b >= B) return;
  float sl = segL[b], sn = segN[b];
  float la0 = LA[((size_t)b * Tdim + t) * 2 + 0] + 10.f * (sl >= 10.f ? 1.f : 0.f);
  float la1 = LA[((size_t)b * Tdim + t) * 2 + 1] + 10.f * (sn >= 5.f ? 1.f : 0.f);
  float u0 = U[((size_t)b * Tdim + t) * 2 + 0];
  float u1 = U[((size_t)b * Tdim + t) * 2 + 1];
  u0 = fminf(fmaxf(u0, 1e-8f), 1.f - 1e-8f);
  u1 = fminf(fmaxf(u1, 1e-8f), 1.f - 1e-8f);
  float g0 = -__logf(-__logf(u0));
  float g1 = -__logf(-__logf(u1));
  float rd = (la0 + g0 >= la1 + g1) ? 1.f : 0.f;  // argmax ties -> index 0
  readv[b] = rd;
  segL[b] = (1.f - rd) * (sl + 1.f);
  segN[b] = sn + rd;
}

__global__ void init_k(float* absB, const float* initb, float* absS, float* obsB,
    float* obsS, float* segL, float* segN, int B, int AB, int AS, int OB, int OS)
{
  int i = blockIdx.x * 256 + threadIdx.x;
  if (i < B * AB) absB[i] = initb[i % AB];
  if (i < B * AS) absS[i] = 0.f;
  if (i < B * OB) obsB[i] = 0.f;
  if (i < B * OS) obsS[i] = 0.f;
  if (i < B) { segL[i] = 0.f; segN[i] = 0.f; }
}

__global__ void zero_k(float* p, int n)
{
  int i = blockIdx.x * 256 + threadIdx.x;
  if (i < n) p[i] = 0.f;
}

// concat up to 3 row-sources into dst with stride dstStride; pad cols zeroed.
__global__ __launch_bounds__(256) void concat3_k(float* __restrict__ dst, int dstStride,
    const float* __restrict__ s0, int ld0, int w0,
    const float* __restrict__ s1, int ld1, int w1,
    const float* __restrict__ s2, int ld2, int w2)
{
  const int row = blockIdx.x;
  const int total = w0 + w1 + w2;
  for (int c = threadIdx.x; c < dstStride; c += 256) {
    float v = 0.f;
    if (c < w0)            v = s0[(size_t)row * ld0 + c];
    else if (c < w0 + w1)  v = s1[(size_t)row * ld1 + (c - w0)];
    else if (c < total)    v = s2[(size_t)row * ld2 + (c - w0 - w1)];
    dst[(size_t)row * dstStride + c] = v;
  }
}

__global__ __launch_bounds__(256) void gru_k(const float* __restrict__ parts,
    const float* __restrict__ state, float* __restrict__ out, int n)
{
  const int row = blockIdx.x;
  const size_t base = (size_t)row * 3 * n;
  for (int c = threadIdx.x; c < n; c += 256) {
    float r  = sigmoidf_(parts[base + c]);
    float cc = tanhf(r * parts[base + n + c]);
    float u  = sigmoidf_(parts[base + 2 * n + c] - 1.f);
    out[(size_t)row * n + c] = u * cc + (1.f - u) * state[(size_t)row * n + c];
  }
}

__global__ __launch_bounds__(256) void blend_copy_k(float* __restrict__ state,
    const float* __restrict__ a, const float* __restrict__ bsrc,
    const float* __restrict__ readv, int N, float* __restrict__ copyDst, int copyLd)
{
  const int row = blockIdx.x;
  const float rd = readv[row];
  for (int c = threadIdx.x; c < N; c += 256) {
    float v = rd * a[(size_t)row * N + c] + (1.f - rd) * bsrc[(size_t)row * N + c];
    state[(size_t)row * N + c] = v;
    if (copyDst) copyDst[(size_t)row * copyLd + c] = v;
  }
}

// ms[B,64] = [mean_raw | std_raw]; out = mean + (softplus(std)+0.1)*eps, opt blend
__global__ void latent_k(const float* __restrict__ ms, const float* __restrict__ mb,
    const float* __restrict__ sb, const float* __restrict__ eps, int ldEps,
    const float* __restrict__ readv, float* __restrict__ stateS,
    float* __restrict__ copyDst, int copyLd)
{
  const int row = blockIdx.x;
  const int c = threadIdx.x;
  if (c >= 32) return;
  float m  = ms[(size_t)row * 64 + c] + mb[c];
  float sr = ms[(size_t)row * 64 + 32 + c] + sb[c];
  float sp = (sr > 20.f) ? sr : log1pf(__expf(sr));
  float v  = m + (sp + 0.1f) * eps[(size_t)row * ldEps + c];
  if (readv) {
    float rd = readv[row];
    v = rd * v + (1.f - rd) * stateS[(size_t)row * 32 + c];
  }
  stateS[(size_t)row * 32 + c] = v;
  if (copyDst) copyDst[(size_t)row * copyLd + c] = v;
}

// f32 [N,K] -> bf16 [N,Kp] with zero pad (Kp >= K)
__global__ void cvt_pad_k(const float* __restrict__ src, unsigned short* __restrict__ dst,
                          int N, int K, int Kp)
{
  int i = blockIdx.x * 256 + threadIdx.x;
  if (i >= N * Kp) return;
  int n = i / Kp, k = i % Kp;
  dst[i] = (k < K) ? f2bf(src[(size_t)n * K + k]) : (unsigned short)0;
}

// (O,I,3) f32 -> tap-major (3,O,I) bf16
__global__ void cvtconv_k(const float* __restrict__ src, unsigned short* __restrict__ dst,
                          int O, int I)
{
  int i = blockIdx.x * 256 + threadIdx.x;
  int n = O * I * 3;
  if (i >= n) return;
  int o = i / (I * 3);
  int rem = i % (I * 3);
  int ii = rem / 3, s = rem % 3;
  dst[(size_t)s * O * I + (size_t)o * I + ii] = f2bf(src[i]);
}

// ---------------------------------------------------------------------------
extern "C" void kernel_launch(void* const* d_in, const int* in_sizes, int n_in,
                              void* d_out, int out_size, void* d_ws, size_t ws_size,
                              hipStream_t stream)
{
  (void)in_sizes; (void)n_in; (void)out_size; (void)ws_size;
  const int cB = 256, cT = 64, cE = 1024, cA = 32, cH = 512;
  const int cAB = 512, cAS = 32, cOB = 512, cOS = 32, cAF = 544;
  const int cAFp = 576;     // 544 padded to 64
  const int cPO  = 2080, cPOp = 2112;  // po trunk K padded to 64
  const int BT = cB * cT;

  const float* embed   = (const float*)d_in[0];
  const float* action  = (const float*)d_in[1];
  const float* noiseU  = (const float*)d_in[2];
  const float* absEps  = (const float*)d_in[3];
  const float* obsEps  = (const float*)d_in[4];
  const float* pb_conv0 = (const float*)d_in[5];
  const float* pb_bn0_g = (const float*)d_in[6];
  const float* pb_bn0_b = (const float*)d_in[7];
  const float* pb_conv1 = (const float*)d_in[8];
  const float* pb_bn1_g = (const float*)d_in[9];
  const float* pb_bn1_b = (const float*)d_in[10];
  const float* pb_out_w = (const float*)d_in[11];
  const float* pb_out_b = (const float*)d_in[12];
  const float* ai_w = (const float*)d_in[13];
  const float* ai_g = (const float*)d_in[14];
  const float* ai_b = (const float*)d_in[15];
  const float* ag_w = (const float*)d_in[16];
  const float* ag_g = (const float*)d_in[17];
  const float* ag_b = (const float*)d_in[18];
  const float* pa_fw = (const float*)d_in[19];
  const float* pa_fg = (const float*)d_in[20];
  const float* pa_fb = (const float*)d_in[21];
  const float* pa_mw = (const float*)d_in[22];
  const float* pa_mb = (const float*)d_in[23];
  const float* pa_sw = (const float*)d_in[24];
  const float* pa_sb = (const float*)d_in[25];
  const float* io_w = (const float*)d_in[26];
  const float* io_g = (const float*)d_in[27];
  const float* io_b = (const float*)d_in[28];
  const float* oi_w = (const float*)d_in[29];
  const float* oi_g = (const float*)d_in[30];
  const float* oi_b = (const float*)d_in[31];
  const float* og_w = (const float*)d_in[32];
  const float* og_g = (const float*)d_in[33];
  const float* og_b = (const float*)d_in[34];
  const float* po_fw = (const float*)d_in[35];
  const float* po_fg = (const float*)d_in[36];
  const float* po_fb = (const float*)d_in[37];
  const float* po_mw = (const float*)d_in[38];
  const float* po_mb = (const float*)d_in[39];
  const float* po_sw = (const float*)d_in[40];
  const float* po_sb = (const float*)d_in[41];
  const float* init_ab = (const float*)d_in[42];
  float* out = (float*)d_out;

  char* wsp = (char*)d_ws;
  size_t off = 0;
  auto allocF = [&](size_t n) -> float* {
    float* p = (float*)(wsp + off); off += ((n * sizeof(float) + 255) / 256) * 256; return p; };
  auto allocU = [&](size_t n) -> unsigned short* {
    unsigned short* p = (unsigned short*)(wsp + off); off += ((n * 2 + 255) / 256) * 256; return p; };

  unsigned short* wc0   = allocU((size_t)3 * cH * cE);
  unsigned short* wc1   = allocU((size_t)3 * cH * cH);
  unsigned short* wai   = allocU((size_t)cH * 64);
  unsigned short* wag   = allocU((size_t)1536 * 1024);
  unsigned short* wpafw = allocU((size_t)cH * 1536);
  unsigned short* wpams = allocU((size_t)64 * cH);
  unsigned short* wio   = allocU((size_t)cOB * cAFp);   // padded K
  unsigned short* woi   = allocU((size_t)cH * 576);
  unsigned short* wog   = allocU((size_t)1536 * 1024);
  unsigned short* wpofw = allocU((size_t)cH * cPOp);    // padded K
  unsigned short* wpoms = allocU((size_t)64 * cH);

  float* Y0   = allocF((size_t)BT * cH);
  float* Y1   = allocF((size_t)BT * cH);
  float* LA   = allocF((size_t)BT * 2);
  float* bnm  = allocF(cH);
  float* bnv  = allocF(cH);
  float* absB = allocF((size_t)cB * cAB);
  float* absS = allocF((size_t)cB * cAS);
  float* obsB = allocF((size_t)cB * cOB);
  float* obsS = allocF((size_t)cB * cOS);
  float* segL = allocF(cB);
  float* segN = allocF(cB);
  float* readv = allocF(cB);
  float* cat1 = allocF((size_t)cB * 64);
  float* ain  = allocF((size_t)cB * cH);
  float* cat2 = allocF((size_t)cB * 1024);
  float* parts = allocF((size_t)cB * 1536);
  float* absBn = allocF((size_t)cB * cAB);
  float* cat3 = allocF((size_t)cB * 1536);
  float* hbuf = allocF((size_t)cB * cH);
  float* msbuf = allocF((size_t)cB * 64);
  float* absFeat = allocF((size_t)cB * cAFp);   // padded stride 576
  float* obInit = allocF((size_t)cB * cOB);
  float* cat4 = allocF((size_t)cB * 576);
  float* oin  = allocF((size_t)cB * cH);
  float* cat5 = allocF((size_t)cB * 1024);
  float* obUpd = allocF((size_t)cB * cOB);
  float* cat6 = allocF((size_t)cB * cPOp);      // padded stride 2112

  auto cdiv = [](int a, int b) { return (a + b - 1) / b; };

  // ---- weight conversion to bf16 (once per launch, deterministic) ----
  cvtconv_k<<<cdiv(3 * cH * cE, 256), 256, 0, stream>>>(pb_conv0, wc0, cH, cE);
  cvtconv_k<<<cdiv(3 * cH * cH, 256), 256, 0, stream>>>(pb_conv1, wc1, cH, cH);
  cvt_pad_k<<<cdiv(cH * 64, 256), 256, 0, stream>>>(ai_w, wai, cH, 64, 64);
  cvt_pad_k<<<cdiv(1536 * 1024, 256), 256, 0, stream>>>(ag_w, wag, 1536, 1024, 1024);
  cvt_pad_k<<<cdiv(cH * 1536, 256), 256, 0, stream>>>(pa_fw, wpafw, cH, 1536, 1536);
  cvt_pad_k<<<cdiv(32 * cH, 256), 256, 0, stream>>>(pa_mw, wpams, 32, cH, cH);
  cvt_pad_k<<<cdiv(32 * cH, 256), 256, 0, stream>>>(pa_sw, wpams + (size_t)32 * cH, 32, cH, cH);
  cvt_pad_k<<<cdiv(cOB * cAFp, 256), 256, 0, stream>>>(io_w, wio, cOB, cAF, cAFp);
  cvt_pad_k<<<cdiv(cH * 576, 256), 256, 0, stream>>>(oi_w, woi, cH, 576, 576);
  cvt_pad_k<<<cdiv(1536 * 1024, 256), 256, 0, stream>>>(og_w, wog, 1536, 1024, 1024);
  cvt_pad_k<<<cdiv(cH * cPOp, 256), 256, 0, stream>>>(po_fw, wpofw, cH, cPO, cPOp);
  cvt_pad_k<<<cdiv(32 * cH, 256), 256, 0, stream>>>(po_mw, wpoms, 32, cH, cH);
  cvt_pad_k<<<cdiv(32 * cH, 256), 256, 0, stream>>>(po_sw, wpoms + (size_t)32 * cH, 32, cH, cH);

  auto gemm = [&](const float* X, const unsigned short* Wb, float* Yo,
                  int M, int N, int K, int ldx, int ntaps, int Tdim) {
    dim3 g(M / 16, cdiv(N, 512));
    gemm_bf16_k<<<g, 256, 0, stream>>>(X, Wb, Yo, M, N, K, ldx, ntaps, Tdim);
  };
  auto ln = [&](float* Xio, const float* g, const float* b, int N, int act) {
    ln_act_k<<<cB, 256, 0, stream>>>(Xio, Xio, g, b, N, act);
  };

  // ---- conv preamble: conv3 -> BN -> SiLU, x2, then 2-logit head ----
  gemm(embed, wc0, Y0, BT, cH, cE, cE, 3, cT);
  bn_stats_k<<<cH, 256, 0, stream>>>(Y0, bnm, bnv, BT, cH);
  bn_silu_k<<<cdiv(BT * cH, 256), 256, 0, stream>>>(Y0, bnm, bnv, pb_bn0_g, pb_bn0_b, BT * cH, cH);
  gemm(Y0, wc1, Y1, BT, cH, cH, cH, 3, cT);
  bn_stats_k<<<cH, 256, 0, stream>>>(Y1, bnm, bnv, BT, cH);
  bn_silu_k<<<cdiv(BT * cH, 256), 256, 0, stream>>>(Y1, bnm, bnv, pb_bn1_g, pb_bn1_b, BT * cH, cH);
  pbout_k<<<BT, 128, 0, stream>>>(Y1, pb_out_w, pb_out_b, LA, cT, cH);

  init_k<<<cdiv(cB * cAB, 256), 256, 0, stream>>>(absB, init_ab, absS, obsB, obsS,
                                                  segL, segN, cB, cAB, cAS, cOB, cOS);
  zero_k<<<cdiv(cB * cAFp, 256), 256, 0, stream>>>(absFeat, cB * cAFp);  // keep pad cols zero

  // ---- sequential scan over T ----
  for (int t = 0; t < cT; ++t) {
    gate_k<<<1, 256, 0, stream>>>(LA, noiseU, segL, segN, readv, t, cT, cB);

    concat3_k<<<cB, 256, 0, stream>>>(cat1, 64, absS, cAS, cAS,
                                      action + (size_t)t * cA, cT * cA, cA, absS, 0, 0);
    gemm(cat1, wai, ain, cB, cH, 64, 64, 1, 1);
    ln(ain, ai_g, ai_b, cH, 1);

    concat3_k<<<cB, 256, 0, stream>>>(cat2, 1024, ain, cH, cH, absB, cAB, cAB, ain, 0, 0);
    gemm(cat2, wag, parts, cB, 1536, 1024, 1024, 1, 1);
    ln(parts, ag_g, ag_b, 1536, 0);
    gru_k<<<cB, 256, 0, stream>>>(parts, absB, absBn, cAB);

    concat3_k<<<cB, 256, 0, stream>>>(cat3, 1536, absBn, cAB, cAB,
                                      embed + (size_t)t * cE, cT * cE, cE, absBn, 0, 0);
    gemm(cat3, wpafw, hbuf, cB, cH, 1536, 1536, 1, 1);
    ln(hbuf, pa_fg, pa_fb, cH, 1);
    gemm(hbuf, wpams, msbuf, cB, 64, cH, cH, 1, 1);

    blend_copy_k<<<cB, 256, 0, stream>>>(absB, absBn, absB, readv, cAB, absFeat, cAFp);
    latent_k<<<cB, 32, 0, stream>>>(msbuf, pa_mb, pa_sb,
                                    absEps + (size_t)t * cAS, cT * cAS,
                                    readv, absS, absFeat + cAB, cAFp);

    gemm(absFeat, wio, obInit, cB, cOB, cAFp, cAFp, 1, 1);  // zero-padded K
    ln(obInit, io_g, io_b, cOB, 1);

    concat3_k<<<cB, 256, 0, stream>>>(cat4, 576, obsS, cOS, cOS,
                                      absFeat, cAFp, cAF, obsS, 0, 0);
    gemm(cat4, woi, oin, cB, cH, 576, 576, 1, 1);
    ln(oin, oi_g, oi_b, cH, 1);

    concat3_k<<<cB, 256, 0, stream>>>(cat5, 1024, oin, cH, cH, obsB, cOB, cOB, oin, 0, 0);
    gemm(cat5, wog, parts, cB, 1536, 1024, 1024, 1, 1);
    ln(parts, og_g, og_b, 1536, 0);
    gru_k<<<cB, 256, 0, stream>>>(parts, obsB, obUpd, cOB);

    blend_copy_k<<<cB, 256, 0, stream>>>(obsB, obInit, obUpd, readv, cOB,
                                         out + (size_t)t * cAF, cT * cAF);

    concat3_k<<<cB, 256, 0, stream>>>(cat6, cPOp, obsB, cOB, cOB,
                                      absFeat, cAFp, cAF,
                                      embed + (size_t)t * cE, cT * cE, cE);
    gemm(cat6, wpofw, hbuf, cB, cH, cPOp, cPOp, 1, 1);      // zero-padded K
    ln(hbuf, po_fg, po_fb, cH, 1);
    gemm(hbuf, wpoms, msbuf, cB, 64, cH, cH, 1, 1);
    latent_k<<<cB, 32, 0, stream>>>(msbuf, po_mb, po_sb,
                                    obsEps + (size_t)t * cOS, cT * cOS,
                                    nullptr, obsS, out + (size_t)t * cAF + cOB, cT * cAF);
  }
}
